// GAT_Numbering_Corrector_V2_6828998000941
// MI455X (gfx1250) — compile-verified, matrix operand order checked by reading, not run on hardware
//
#include <hip/hip_runtime.h>

typedef __attribute__((ext_vector_type(16))) __bf16 v16bf;
typedef __attribute__((ext_vector_type(8)))  __bf16 v8bf;
typedef __attribute__((ext_vector_type(8)))  float  v8f;

#define HEADS 4
#define BM 64
#define BN 64
#define BK 32
#define LDS_STRIDE 40   // bf16 elems per LDS row (64B data + 16B pad, 16B-aligned rows)

// ---------------- utility kernels ----------------

__global__ __launch_bounds__(256) void fill_f32(float* p, float v, long long n) {
  long long i = (long long)blockIdx.x * blockDim.x + threadIdx.x;
  if (i < n) p[i] = v;
}

// Wt[n*K + k] = (bf16) W[k*N + n]   (convert + transpose weights once)
__global__ __launch_bounds__(256) void cvt_w_transpose(const float* __restrict__ W,
                                                       __bf16* __restrict__ Wt,
                                                       int K, int N) {
  long long i = (long long)blockIdx.x * blockDim.x + threadIdx.x;
  if (i < (long long)K * N) {
    int k = (int)(i / N), n = (int)(i % N);
    Wt[(long long)n * K + k] = (__bf16)W[i];
  }
}

__global__ __launch_bounds__(256) void build_edges(const int* __restrict__ ei,
                                                   int* __restrict__ src,
                                                   int* __restrict__ dst,
                                                   int E, int Nn) {
  int i = blockIdx.x * blockDim.x + threadIdx.x;
  if (i < E) { src[i] = ei[i]; dst[i] = ei[E + i]; }
  else if (i < E + Nn) { int v = i - E; src[i] = v; dst[i] = v; }
}

// ---------------- WMMA GEMM: C[M,N] = A[M,K](f32) * Bt[N,K](bf16)^T (+bias) ----------------

__device__ inline v8bf cvt8(float4 a, float4 b) {
  v8bf r;
  r[0] = (__bf16)a.x; r[1] = (__bf16)a.y; r[2] = (__bf16)a.z; r[3] = (__bf16)a.w;
  r[4] = (__bf16)b.x; r[5] = (__bf16)b.y; r[6] = (__bf16)b.z; r[7] = (__bf16)b.w;
  return r;
}

__global__ __launch_bounds__(128) void gemm_bf16_wmma(
    const float*  __restrict__ A,     // [M,K] f32
    const __bf16* __restrict__ Bt,    // [N,K] bf16 (pre-transposed weights)
    const float*  __restrict__ bias,  // [N] or nullptr
    float*        __restrict__ C,     // [M,N]
    int M, int K, int N) {
  __shared__ __bf16 sA[BM * LDS_STRIDE];
  __shared__ __bf16 sB[BN * LDS_STRIDE];

  const int tid  = threadIdx.x;
  const int lane = tid & 31;
  const int wave = tid >> 5;
  const int mlan = lane & 15;
  const int half = lane >> 4;        // 0: K 0-7 & 16-23, 1: K 8-15 & 24-31 (ISA 16-bit A layout)
  const int c0   = half * 8;
  const int rbase = wave * 16;       // 4 waves -> rows 0..63

  const int m0 = blockIdx.x * BM;
  const int n0 = blockIdx.y * BN;

  v8f acc[4] = {};

  const int srow = tid >> 1;         // staging: 0..63
  const int skk  = (tid & 1) * 16;   // 0 or 16

  // block-uniform fast-path predicates (scalar branches)
  const bool kAligned = ((K & (BK - 1)) == 0);    // every K-tile full & 16B-aligned rows
  const bool aFull = kAligned && (m0 + BM <= M);
  const bool bFull = kAligned && (n0 + BN <= N);

  __bf16* sAp = &sA[srow * LDS_STRIDE + skk];
  __bf16* sBp = &sB[srow * LDS_STRIDE + skk];

  for (int k0 = 0; k0 < K; k0 += BK) {
    // ---------------- stage A tile (f32 -> bf16) ----------------
    if (aFull) {
      const float4* ap4 = (const float4*)(A + (long long)(m0 + srow) * K + (k0 + skk));
      float4 f0 = ap4[0], f1 = ap4[1], f2 = ap4[2], f3 = ap4[3];
      if (k0 + BK < K) __builtin_prefetch(ap4 + BK / 4, 0, 1);   // next K-tile
      *(v8bf*)(sAp)     = cvt8(f0, f1);
      *(v8bf*)(sAp + 8) = cvt8(f2, f3);
    } else {
      long long gr  = (long long)m0 + srow;
      long long grc = gr < M ? gr : (long long)(M - 1);
      const float* ap = A + grc * (long long)K;
      const bool rin = gr < M;
      #pragma unroll
      for (int j = 0; j < 16; ++j) {
        int k  = k0 + skk + j;
        int kc = k < K ? k : (K - 1);
        float v = ap[kc];                       // always in-range address
        v = (rin && k < K) ? v : 0.f;           // branchless zero-fill
        sAp[j] = (__bf16)v;
      }
    }
    // ---------------- stage B tile (bf16, contiguous along K) ----------------
    if (bFull) {
      const v8bf* bp8 = (const v8bf*)(Bt + (long long)(n0 + srow) * K + (k0 + skk));
      v8bf b0 = bp8[0], b1 = bp8[1];
      if (k0 + BK < K) __builtin_prefetch(bp8 + BK / 8, 0, 1);
      *(v8bf*)(sBp)     = b0;
      *(v8bf*)(sBp + 8) = b1;
    } else {
      long long gn  = (long long)n0 + srow;
      long long gnc = gn < N ? gn : (long long)(N - 1);
      const __bf16* bp = Bt + gnc * (long long)K;
      const bool nin = gn < N;
      #pragma unroll
      for (int j = 0; j < 16; ++j) {
        int k  = k0 + skk + j;
        int kc = k < K ? k : (K - 1);
        __bf16 v = bp[kc];
        v = (nin && k < K) ? v : (__bf16)0.f;
        sBp[j] = v;
      }
    }
    __syncthreads();

    union { v16bf v; v8bf h[2]; } fa;
    const __bf16* arow = &sA[(rbase + mlan) * LDS_STRIDE];
    fa.h[0] = *(const v8bf*)(arow + c0);
    fa.h[1] = *(const v8bf*)(arow + c0 + 16);
    #pragma unroll
    for (int nt = 0; nt < 4; ++nt) {
      union { v16bf v; v8bf h[2]; } fb;
      const __bf16* brow = &sB[(nt * 16 + mlan) * LDS_STRIDE];
      fb.h[0] = *(const v8bf*)(brow + c0);
      fb.h[1] = *(const v8bf*)(brow + c0 + 16);
      acc[nt] = __builtin_amdgcn_wmma_f32_16x16x32_bf16(
          false, fa.v, false, fb.v, (short)0, acc[nt], false, false);
    }
    __syncthreads();
  }

  // D layout (f32 16x16): VGPR r, lanes 0-15 -> M=r, lanes 16-31 -> M=8+r; N=lane&15
  #pragma unroll
  for (int nt = 0; nt < 4; ++nt) {
    int gc = n0 + nt * 16 + mlan;
    #pragma unroll
    for (int r = 0; r < 8; ++r) {
      int gr = m0 + rbase + half * 8 + r;
      if (gr < M && gc < N) {
        float v = acc[nt][r];
        if (bias) v += bias[gc];
        C[(long long)gr * N + gc] = v;
      }
    }
  }
}

// ---------------- edge-phase kernels ----------------

__device__ inline void atomicMaxF(float* addr, float val) {
  // order-preserving int trick (values finite)
  if (val >= 0.f) atomicMax((int*)addr, __float_as_int(val));
  else            atomicMin((unsigned int*)addr, __float_as_uint(val));
}

// one wave32 per (edge, head): logits[e,h] = sum_c att[h,c]*leaky(xl[src]+xr[dst]); atomic max per (dst,h)
__global__ __launch_bounds__(256) void edge_logits_k(
    const float* __restrict__ xl, const float* __restrict__ xr,
    const float* __restrict__ att, const int* __restrict__ src,
    const int* __restrict__ dst, float* __restrict__ logits,
    float* __restrict__ maxbuf, int EH, int C, int d2) {
  int w = blockIdx.x * (blockDim.x >> 5) + (threadIdx.x >> 5);
  if (w >= EH) return;
  int lane = threadIdx.x & 31;
  int e = w >> 2, h = w & 3;
  int s = src[e], d = dst[e];
  const float* pl = xl + (long long)s * d2 + h * C;
  const float* pr = xr + (long long)d * d2 + h * C;
  const float* pa = att + h * C;
  float sum = 0.f;
  if ((C & 127) == 0) {
    // 32 lanes x float4 = 128 channels per sweep (one b128 gather per matrix per wave)
    for (int cb = lane * 4; cb < C; cb += 128) {
      float4 a = *(const float4*)(pl + cb);
      float4 b = *(const float4*)(pr + cb);
      float4 wv = *(const float4*)(pa + cb);
      float v0 = a.x + b.x, v1 = a.y + b.y, v2 = a.z + b.z, v3 = a.w + b.w;
      v0 = v0 > 0.f ? v0 : 0.2f * v0;
      v1 = v1 > 0.f ? v1 : 0.2f * v1;
      v2 = v2 > 0.f ? v2 : 0.2f * v2;
      v3 = v3 > 0.f ? v3 : 0.2f * v3;
      sum += v0 * wv.x + v1 * wv.y + v2 * wv.z + v3 * wv.w;
    }
  } else {
    for (int c = lane; c < C; c += 32) {
      float v = pl[c] + pr[c];
      v = v > 0.f ? v : 0.2f * v;
      sum += v * pa[c];
    }
  }
  #pragma unroll
  for (int off = 16; off > 0; off >>= 1) sum += __shfl_down(sum, off, 32);
  if (lane == 0) {
    logits[w] = sum;
    atomicMaxF(maxbuf + (long long)d * HEADS + h, sum);
  }
}

__global__ __launch_bounds__(256) void edge_exp_k(
    const float* __restrict__ logits, const int* __restrict__ dst,
    const float* __restrict__ maxbuf, float* __restrict__ pbuf,
    float* __restrict__ denom, int EH) {
  int i = blockIdx.x * blockDim.x + threadIdx.x;
  if (i >= EH) return;
  int e = i >> 2, h = i & 3;
  int d = dst[e];
  float m = maxbuf[(long long)d * HEADS + h];   // every node has a self-loop -> finite
  float p = __expf(logits[i] - m);
  pbuf[i] = p;
  atomicAdd(denom + (long long)d * HEADS + h, p);
}

// one wave32 per edge: hacc[dst] += (p/denom) * xl[src]
__global__ __launch_bounds__(256) void edge_aggr_k(
    const float* __restrict__ xl, const float* __restrict__ pbuf,
    const float* __restrict__ denom, const int* __restrict__ src,
    const int* __restrict__ dst, float* __restrict__ hacc,
    int E, int C, int d2) {
  int w = blockIdx.x * (blockDim.x >> 5) + (threadIdx.x >> 5);
  if (w >= E) return;
  int lane = threadIdx.x & 31;
  int s = src[w], d = dst[w];
  float alpha[HEADS];
  #pragma unroll
  for (int h = 0; h < HEADS; ++h)
    alpha[h] = pbuf[(long long)w * HEADS + h] / denom[(long long)d * HEADS + h];
  const float* pl = xl + (long long)s * d2;
  float* po = hacc + (long long)d * d2;
  if ((C & 127) == 0 && (d2 & 127) == 0) {
    for (int cb = lane * 4; cb < d2; cb += 128) {
      float4 v = *(const float4*)(pl + cb);
      float al = alpha[cb / C];   // cb..cb+3 in same head (C multiple of 4)
      atomicAdd(po + cb + 0, al * v.x);
      atomicAdd(po + cb + 1, al * v.y);
      atomicAdd(po + cb + 2, al * v.z);
      atomicAdd(po + cb + 3, al * v.w);
    }
  } else {
    for (int c = lane; c < d2; c += 32)
      atomicAdd(po + c, alpha[c / C] * pl[c]);
  }
}

__global__ __launch_bounds__(256) void bias_act_k(
    const float* __restrict__ acc, const float* __restrict__ bias,
    float* __restrict__ out, int d2, long long n, int do_elu) {
  long long i4 = ((long long)blockIdx.x * blockDim.x + threadIdx.x) * 4;
  if (i4 + 3 < n) {
    float4 v = *(const float4*)(acc + i4);
    int cb = (int)(i4 % d2);      // d2 multiple of 4 -> cb..cb+3 valid
    v.x += bias[cb + 0]; v.y += bias[cb + 1]; v.z += bias[cb + 2]; v.w += bias[cb + 3];
    if (do_elu) {
      v.x = v.x > 0.f ? v.x : expm1f(v.x);
      v.y = v.y > 0.f ? v.y : expm1f(v.y);
      v.z = v.z > 0.f ? v.z : expm1f(v.z);
      v.w = v.w > 0.f ? v.w : expm1f(v.w);
    }
    *(float4*)(out + i4) = v;
  } else {
    for (long long i = i4; i < n; ++i) {
      float v = acc[i] + bias[(int)(i % d2)];
      if (do_elu) v = v > 0.f ? v : expm1f(v);
      out[i] = v;
    }
  }
}

// ---------------- host orchestration ----------------

extern "C" void kernel_launch(void* const* d_in, const int* in_sizes, int n_in,
                              void* d_out, int out_size, void* d_ws, size_t ws_size,
                              hipStream_t stream) {
  const float* x    = (const float*)d_in[0];
  const int*   ei   = (const int*)d_in[1];
  const float* W1l  = (const float*)d_in[2];
  const float* W1r  = (const float*)d_in[3];
  const float* att1 = (const float*)d_in[4];
  const float* b1   = (const float*)d_in[5];
  const float* W2l  = (const float*)d_in[6];
  const float* W2r  = (const float*)d_in[7];
  const float* att2 = (const float*)d_in[8];
  const float* b2   = (const float*)d_in[9];
  const float* Wc   = (const float*)d_in[10];
  const float* bc   = (const float*)d_in[11];

  const int Fin  = 55;
  const int d2   = in_sizes[4];          // H*C = 512
  const int C    = d2 / HEADS;           // 128
  const int Nn   = in_sizes[0] / Fin;    // 50000
  const int E    = in_sizes[1] / 2;      // 400000
  const int Ep   = E + Nn;               // + self loops
  const int ncls = in_sizes[10] / d2;    // 49
  (void)n_in; (void)out_size; (void)ws_size;

  // carve workspace
  char* wp = (char*)d_ws;
  auto take = [&](size_t bytes) -> char* {
    char* r = wp; wp += (bytes + 255) & ~(size_t)255; return r;
  };
  float*  xl    = (float*) take((size_t)Nn * d2 * 4);
  float*  xr    = (float*) take((size_t)Nn * d2 * 4);
  float*  hacc  = (float*) take((size_t)Nn * d2 * 4);
  float*  hcur  = (float*) take((size_t)Nn * d2 * 4);
  float*  logit = (float*) take((size_t)Ep * HEADS * 4);
  float*  pbuf  = (float*) take((size_t)Ep * HEADS * 4);
  float*  maxb  = (float*) take((size_t)Nn * HEADS * 4);
  float*  den   = (float*) take((size_t)Nn * HEADS * 4);
  int*    srcb  = (int*)   take((size_t)Ep * 4);
  int*    dstb  = (int*)   take((size_t)Ep * 4);
  __bf16* w1lT  = (__bf16*)take((size_t)Fin * d2 * 2);
  __bf16* w1rT  = (__bf16*)take((size_t)Fin * d2 * 2);
  __bf16* w2lT  = (__bf16*)take((size_t)d2 * d2 * 2);
  __bf16* w2rT  = (__bf16*)take((size_t)d2 * d2 * 2);
  __bf16* wcT   = (__bf16*)take((size_t)d2 * ncls * 2);

  auto cvt = [&](const float* W, __bf16* Wt, int K, int N) {
    long long n = (long long)K * N;
    cvt_w_transpose<<<dim3((unsigned)((n + 255) / 256)), 256, 0, stream>>>(W, Wt, K, N);
  };
  cvt(W1l, w1lT, Fin, d2);
  cvt(W1r, w1rT, Fin, d2);
  cvt(W2l, w2lT, d2, d2);
  cvt(W2r, w2rT, d2, d2);
  cvt(Wc,  wcT,  d2, ncls);

  build_edges<<<dim3((unsigned)((Ep + 255) / 256)), 256, 0, stream>>>(ei, srcb, dstb, E, Nn);

  const int EH = Ep * HEADS;
  auto run_gemm = [&](const float* A, const __bf16* Bt, const float* bias, float* Cc,
                      int M, int K, int N) {
    dim3 g((unsigned)((M + BM - 1) / BM), (unsigned)((N + BN - 1) / BN));
    gemm_bf16_wmma<<<g, 128, 0, stream>>>(A, Bt, bias, Cc, M, K, N);
  };
  auto run_layer = [&](const float* hin, int K, const __bf16* WlT, const __bf16* WrT,
                       const float* att, const float* bias, float* hout, int do_elu) {
    run_gemm(hin, WlT, nullptr, xl, Nn, K, d2);
    run_gemm(hin, WrT, nullptr, xr, Nn, K, d2);
    long long nh = (long long)Nn * HEADS;
    long long nd = (long long)Nn * d2;
    fill_f32<<<dim3((unsigned)((nh + 255) / 256)), 256, 0, stream>>>(maxb, -1e30f, nh);
    fill_f32<<<dim3((unsigned)((nh + 255) / 256)), 256, 0, stream>>>(den, 0.f, nh);
    fill_f32<<<dim3((unsigned)((nd + 255) / 256)), 256, 0, stream>>>(hacc, 0.f, nd);
    edge_logits_k<<<dim3((unsigned)((EH + 7) / 8)), 256, 0, stream>>>(
        xl, xr, att, srcb, dstb, logit, maxb, EH, C, d2);
    edge_exp_k<<<dim3((unsigned)((EH + 255) / 256)), 256, 0, stream>>>(
        logit, dstb, maxb, pbuf, den, EH);
    edge_aggr_k<<<dim3((unsigned)((Ep + 7) / 8)), 256, 0, stream>>>(
        xl, pbuf, den, srcb, dstb, hacc, Ep, C, d2);
    bias_act_k<<<dim3((unsigned)((nd / 4 + 255) / 256)), 256, 0, stream>>>(
        hacc, bias, hout, d2, nd, do_elu);
  };

  run_layer(x,    Fin, w1lT, w1rT, att1, b1, hcur, 1);
  run_layer(hcur, d2,  w2lT, w2rT, att2, b2, hcur, 1);
  run_layer(hcur, d2,  w2lT, w2rT, att2, b2, hcur, 1);  // conv2 applied twice per reference

  run_gemm(hcur, wcT, bc, (float*)d_out, Nn, d2, ncls);
}
